// VectorQuantizer_67138928771521
// MI455X (gfx1250) — matrix operand order repaired and checked by reading, not compiled, not run on hardware
//
#include <hip/hip_runtime.h>
#include <stdint.h>

// Problem constants (match reference)
#define K_CODES 8192
#define DIM     256
#define N_ROWS  (32 * 1024)
#define NELEM   ((size_t)N_ROWS * DIM)   // 8388608

typedef __attribute__((ext_vector_type(16))) __bf16 v16bf;
typedef __attribute__((ext_vector_type(8)))  float  v8f;

union Frag16 {
    v16bf          v;
    unsigned short h[16];
    uint4          q[2];
};

// float -> bf16 round-to-nearest-even
static __device__ __forceinline__ unsigned short f2bf(float f) {
    uint32_t u = __float_as_uint(f);
    uint32_t r = u + 0x7FFFu + ((u >> 16) & 1u);
    return (unsigned short)(r >> 16);
}

// ---------------------------------------------------------------------------
// Pass 1: embeddings fp32 -> bf16 table + per-code squared norms; zero loss acc
// One wave per code row; 8 rows per 256-thread block.
// ---------------------------------------------------------------------------
__global__ void vq_prep_emb(const float* __restrict__ emb,
                            unsigned short* __restrict__ ebf,
                            float* __restrict__ esq,
                            float* __restrict__ acc) {
    if (blockIdx.x == 0 && threadIdx.x == 0) *acc = 0.0f;
    const int lane = threadIdx.x & 31;
    const int wave = threadIdx.x >> 5;
    const int row  = blockIdx.x * 8 + wave;           // 1024 blocks * 8 = 8192
    const float* r      = emb + (size_t)row * DIM;
    unsigned short* w   = ebf + (size_t)row * DIM;
    float s = 0.0f;
#pragma unroll
    for (int t = 0; t < 8; ++t) {
        int c = t * 32 + lane;
        float v = r[c];
        s = fmaf(v, v, s);
        w[c] = f2bf(v);
    }
#pragma unroll
    for (int m = 16; m >= 1; m >>= 1) s += __shfl_xor(s, m, 32);
    if (lane == 0) esq[row] = s;
}

// ---------------------------------------------------------------------------
// Pass 2: main kernel. 8 waves / block, wave owns 16 rows of x.
// A (x tile, bf16) resident in VGPRs for the whole K sweep; B tiles streamed
// from the bf16 codebook (WGP$/L2-resident). Two independent 4-deep WMMA
// accumulation chains per tile to avoid serializing the matrix pipe.
// ---------------------------------------------------------------------------
__global__ void __launch_bounds__(256)
vq_main(const float* __restrict__ x,
        const float* __restrict__ emb,
        const unsigned short* __restrict__ ebf,
        const float* __restrict__ esq,
        float* __restrict__ qout,
        float* __restrict__ idxout,
        float* __restrict__ acc) {
    __shared__ float s_esq[K_CODES];                  // 32 KB of the 320 KB WGP LDS
    for (int i = threadIdx.x; i < K_CODES; i += 256) s_esq[i] = esq[i];
    __syncthreads();

    const int lane    = threadIdx.x & 31;
    const int wave    = threadIdx.x >> 5;
    const int half    = lane >> 4;                    // 0/1: lane half
    const int l15     = lane & 15;
    const int rowbase = blockIdx.x * 128 + wave * 16;
    const int arow    = rowbase + l15;

    // --- Load A fragments: 16-bit A-matrix 16x32 swizzle (ISA 7.12.2) ---
    // lane<16: K = {d0..d0+7, d0+16..d0+23}; lane>=16: +8 on both chunks.
    Frag16 a[8];
    {
        const float* xr = x + (size_t)arow * DIM;
#pragma unroll
        for (int f = 0; f < 8; ++f) {
            const int c0 = f * 32 + half * 8;
            const int c1 = c0 + 16;
            float tmp[16];
            *(float4*)(tmp + 0)  = *(const float4*)(xr + c0);
            *(float4*)(tmp + 4)  = *(const float4*)(xr + c0 + 4);
            *(float4*)(tmp + 8)  = *(const float4*)(xr + c1);
            *(float4*)(tmp + 12) = *(const float4*)(xr + c1 + 4);
#pragma unroll
            for (int i = 0; i < 16; ++i) a[f].h[i] = f2bf(tmp[i]);
        }
    }

    // Running argmin state. D layout (f32 16x16): lane l, vgpr j holds
    // element (M = j + 8*(l>>4), N = l&15).
    float best[8];
    int   bidx[8];
#pragma unroll
    for (int j = 0; j < 8; ++j) { best[j] = 3.402823466e38f; bidx[j] = 0; }

    for (int kt = 0; kt < K_CODES / 16; ++kt) {
        const int code = kt * 16 + l15;               // column this lane owns
        // B 32x16 bf16: lane = N; lanes<16 hold K {0..7,16..23}, lanes>=16 hold {8..15,24..31}
        const unsigned short* er = ebf + (size_t)code * DIM + half * 8;
        // WGP-scope prefetch (locality 3) two codebook rows ahead: B tiles are shared
        // by all 8 waves of the block, so we want them in WGP$ not just GL2.
        __builtin_prefetch(er + 2 * DIM, 0, 3);

        v8f c0 = {};                                  // even fragments
        v8f c1 = {};                                  // odd fragments
#pragma unroll
        for (int f = 0; f < 8; f += 2) {
            Frag16 b0, b1;
            b0.q[0] = *(const uint4*)(er + f * 32);
            b0.q[1] = *(const uint4*)(er + f * 32 + 16);
            b1.q[0] = *(const uint4*)(er + f * 32 + 32);
            b1.q[1] = *(const uint4*)(er + f * 32 + 48);
            c0 = __builtin_amdgcn_wmma_f32_16x16x32_bf16(
                false, a[f].v, false, b0.v, (short)0, c0, false, false);
            c1 = __builtin_amdgcn_wmma_f32_16x16x32_bf16(
                false, a[f + 1].v, false, b1.v, (short)0, c1, false, false);
        }

        const float e2 = s_esq[code];
#pragma unroll
        for (int j = 0; j < 8; ++j) {
            // score = e2 - 2*(c0+c1); x^2 omitted (row-constant for argmin)
            float score = fmaf(-2.0f, c0[j], fmaf(-2.0f, c1[j], e2));
            if (score < best[j]) { best[j] = score; bidx[j] = code; }
        }
    }

    // Reduce argmin across the 16 columns (lanes within each half).
    // Tie-break: smaller index (matches jnp.argmin first-occurrence).
#pragma unroll
    for (int j = 0; j < 8; ++j) {
#pragma unroll
        for (int m = 1; m <= 8; m <<= 1) {
            float os = __shfl_xor(best[j], m, 32);
            int   oi = __shfl_xor(bidx[j], m, 32);
            if (os < best[j] || (os == best[j] && oi < bidx[j])) {
                best[j] = os; bidx[j] = oi;
            }
        }
    }

    // Gather fp32 codebook rows, write quantized + indices, accumulate MSE.
    float sumsq = 0.0f;
#pragma unroll
    for (int r = 0; r < 16; ++r) {
        const int j = r & 7;
        const int h = r >> 3;
        const int idx = __shfl(bidx[j], h << 4, 32);  // broadcast from owner half
        const int row = rowbase + r;
        if (lane == 0) idxout[row] = (float)idx;
        const float* qsrc = emb + (size_t)idx * DIM;
        const float* xr   = x   + (size_t)row * DIM;
        float*       o    = qout + (size_t)row * DIM;
#pragma unroll
        for (int t = 0; t < 8; ++t) {
            const int c = t * 32 + lane;
            float qv = qsrc[c];
            float d  = xr[c] - qv;
            o[c] = qv;                                // quantized_st == quantized
            sumsq = fmaf(d, d, sumsq);
        }
    }
#pragma unroll
    for (int m = 16; m >= 1; m >>= 1) sumsq += __shfl_xor(sumsq, m, 32);
    if (lane == 0) atomicAdd(acc, sumsq);
}

// ---------------------------------------------------------------------------
// Pass 3: losses from the accumulated sum of squared diffs.
// ---------------------------------------------------------------------------
__global__ void vq_finalize(const float* __restrict__ acc,
                            float* __restrict__ loss) {
    float mean = *acc * (1.0f / (float)NELEM);
    loss[0] = 0.25f * mean;   // commitment
    loss[1] = mean;           // codebook
    loss[2] = 1.25f * mean;   // total
}

extern "C" void kernel_launch(void* const* d_in, const int* in_sizes, int n_in,
                              void* d_out, int out_size, void* d_ws, size_t ws_size,
                              hipStream_t stream) {
    (void)in_sizes; (void)n_in; (void)out_size; (void)ws_size;
    const float* x   = (const float*)d_in[0];   // [32,1024,256] f32
    const float* emb = (const float*)d_in[1];   // [8192,256] f32

    float* qout   = (float*)d_out;              // 8388608 floats
    float* idxout = qout + NELEM;               // 32768 floats
    float* loss   = idxout + N_ROWS;            // 3 floats

    unsigned short* ebf = (unsigned short*)d_ws;                     // 4 MB bf16 codebook
    float* esq = (float*)((char*)d_ws + (size_t)K_CODES * DIM * 2);  // 32 KB norms
    float* acc = esq + K_CODES;                                      // 1 float

    vq_prep_emb<<<K_CODES / 8, 256, 0, stream>>>(emb, ebf, esq, acc);
    vq_main<<<N_ROWS / 128, 256, 0, stream>>>(x, emb, ebf, esq, qout, idxout, acc);
    vq_finalize<<<1, 1, 0, stream>>>(acc, loss);
}
